// QuantumLayer_35253091565612
// MI455X (gfx1250) — matrix operand order repaired and checked by reading, not compiled, not run on hardware
//
#include <hip/hip_runtime.h>
#include <hip/hip_bf16.h>

typedef __attribute__((ext_vector_type(16))) _Float16 v16h;
typedef __attribute__((ext_vector_type(8)))  float    v8f;

// ---------------------------------------------------------------------------
// Kernel A: fold the whole 4-layer StronglyEntanglingLayers circuit (fixed
// weights) into a 32x32 complex unitary. Thread i pushes basis state e_i
// through the circuit; its final amplitudes are column i of U. We emit the
// real 64x32 matrix M = [Re(U); Im(U)] directly in a contiguous-k v_wmma f16 B-operand
// layout: chunk c (N = c*16..c*16+15), lane l: N = c*16 + (l&15),
// K = 16*(l>>4) + elem  (elems ascending within the lane's 16 halves).
// ---------------------------------------------------------------------------
__global__ void __launch_bounds__(32)
build_unitary(const float* __restrict__ w, _Float16* __restrict__ Bbuf) {
    const int i = threadIdx.x;          // column / input basis index (K)
    float re[32], im[32];
#pragma unroll
    for (int j = 0; j < 32; ++j) { re[j] = (j == i) ? 1.f : 0.f; im[j] = 0.f; }

#pragma unroll
    for (int l = 0; l < 4; ++l) {
        // ---- single-qubit Rot(phi,theta,omega) on each wire ----
#pragma unroll
        for (int q = 0; q < 5; ++q) {
            const float phi = w[(l * 5 + q) * 3 + 0];
            const float th  = w[(l * 5 + q) * 3 + 1];
            const float om  = w[(l * 5 + q) * 3 + 2];
            const float ct = __cosf(th * 0.5f), st = __sinf(th * 0.5f);
            const float ap = -0.5f * (phi + om), am = -0.5f * (phi - om);
            const float epr = __cosf(ap), epi = __sinf(ap);
            const float emr = __cosf(am), emi = __sinf(am);
            // U00=ep*c  U01=-conj(em)*s  U10=em*s  U11=conj(ep)*c
            const float u00r =  epr * ct, u00i =  epi * ct;
            const float u01r = -emr * st, u01i =  emi * st;
            const float u10r =  emr * st, u10i =  emi * st;
            const float u11r =  epr * ct, u11i = -epi * ct;
            const int mask = 1 << (4 - q);   // axis 1 = qubit 0 = MSB
#pragma unroll
            for (int j = 0; j < 32; ++j) {
                if (j & mask) continue;
                const int j1 = j | mask;
                const float a0r = re[j],  a0i = im[j];
                const float a1r = re[j1], a1i = im[j1];
                re[j]  = u00r * a0r - u00i * a0i + u01r * a1r - u01i * a1i;
                im[j]  = u00r * a0i + u00i * a0r + u01r * a1i + u01i * a1r;
                re[j1] = u10r * a0r - u10i * a0i + u11r * a1r - u11i * a1i;
                im[j1] = u10r * a0i + u10i * a0r + u11r * a1i + u11i * a1r;
            }
        }
        // ---- ring of CNOTs, range r = (l % 4) + 1 ----
        const int r = (l % 4) + 1;
#pragma unroll
        for (int q = 0; q < 5; ++q) {
            const int t  = (q + r) % 5;
            const int cm = 1 << (4 - q), tm = 1 << (4 - t);
#pragma unroll
            for (int j = 0; j < 32; ++j) {
                if ((j & cm) && !(j & tm)) {
                    const int j1 = j | tm;
                    float tr = re[j]; re[j] = re[j1]; re[j1] = tr;
                    float ti = im[j]; im[j] = im[j1]; im[j1] = ti;
                }
            }
        }
    }

    // ---- scatter column i into WMMA B layout (f16) ----
    const int khi = i >> 4, klo = i & 15;
#pragma unroll
    for (int c = 0; c < 4; ++c) {
#pragma unroll
        for (int n = 0; n < 16; ++n) {
            const int N = c * 16 + n;                 // row of M (output idx)
            const float val = (N < 32) ? re[N] : im[N - 32];
            Bbuf[(c * 32 + 16 * khi + n) * 16 + klo] = (_Float16)val;
        }
    }
}

// ---------------------------------------------------------------------------
// Kernel B: memory-bound main pass. 256 threads = 8 waves; each wave owns a
// 16-sample tile; block handles 128 consecutive samples.
//   Stage 1: coalesced b128 loads -> LDS -> per-sample rsqrt normalize ->
//            f16 A-operand layout -> 4x v_wmma_f32_16x16x32_f16 (K=32).
//   Stage 2: squares -> probs scattered back into the A-tile (transpose) ->
//            5th WMMA against an in-register +/-1 sign matrix computes all 5
//            PauliZ expvals for 16 samples at once -> staged + coalesced out.
// A-operand layout (ISA 7.12.2, 16-bit A 16x32): lane = 16*khalf + m;
//   elems 0-7  -> K = 8*khalf + e ; elems 8-15 -> K = 16 + 8*khalf + (e-8)
// ---------------------------------------------------------------------------
__global__ void __launch_bounds__(256)
qsim_main(const float* __restrict__ in, const _Float16* __restrict__ Bbuf,
          float* __restrict__ out) {
    __shared__ float raw[128 * 30];                         // 15360 B (reused)
    __shared__ __align__(32) _Float16 Atile[8][32][16];     //  8192 B

    const int tid = threadIdx.x;

    // ---- stage inputs: 960 float4 per block, fully coalesced ----
    {
        const float4* in4 = (const float4*)(in + (long long)blockIdx.x * (128 * 30));
        float4* raw4 = (float4*)raw;
#pragma unroll
        for (int i = 0; i < 4; ++i) {
            const int idx = tid + i * 256;
            if (idx < 960) raw4[idx] = in4[idx];
        }
    }
    __syncthreads();

    const int wave = tid >> 5, lane = tid & 31;

    // ---- normalize one sample per lane (lanes 0-15), emit f16 A layout ----
    if (lane < 16) {
        const int s = wave * 16 + lane;
        float v[30]; float ss = 0.f;
#pragma unroll
        for (int k = 0; k < 30; ++k) { v[k] = raw[s * 30 + k]; ss += v[k] * v[k]; }
        const float rn = __frsqrt_rn(ss);
#pragma unroll
        for (int k = 0; k < 30; ++k) {
            const int kh = k >> 4;
            const int e  = k & 15;
            Atile[wave][16 * kh + lane][e] = (_Float16)(v[k] * rn);
        }
        Atile[wave][16 + lane][14] = (_Float16)0.f;         // pad K=30,31
        Atile[wave][16 + lane][15] = (_Float16)0.f;
    }
    __syncthreads();

    const v16h a  = *(const v16h*)(&Atile[wave][lane][0]);
    const v16h b0 = *(const v16h*)(Bbuf + (0 * 32 + lane) * 16);
    const v16h b1 = *(const v16h*)(Bbuf + (1 * 32 + lane) * 16);
    const v16h b2 = *(const v16h*)(Bbuf + (2 * 32 + lane) * 16);
    const v16h b3 = *(const v16h*)(Bbuf + (3 * 32 + lane) * 16);

    v8f z = {};
    // chunk0: Re(amp n)   chunk1: Re(amp n+16)
    // chunk2: Im(amp n)   chunk3: Im(amp n+16)      (n = lane & 15)
    v8f acc0 = __builtin_amdgcn_wmma_f32_16x16x32_f16(false, a, false, b0, (short)0, z, false, false);
    v8f acc1 = __builtin_amdgcn_wmma_f32_16x16x32_f16(false, a, false, b1, (short)0, z, false, false);
    v8f acc2 = __builtin_amdgcn_wmma_f32_16x16x32_f16(false, a, false, b2, (short)0, z, false, false);
    v8f acc3 = __builtin_amdgcn_wmma_f32_16x16x32_f16(false, a, false, b3, (short)0, z, false, false);

    __syncthreads();    // everyone done reading Atile phase-1 contents

    // ---- transpose probs into A layout: p[K=n] / p[K=n+16] share one row ----
    const int n  = lane & 15;                 // amplitude index low 4 bits
    const int mh = (lane >> 4) * 8;           // D layout: lanes16-31 hold M=8..15
    const int prow = 16 * ((n >> 3) & 1);     // khalf for K=n (and K=n+16)
#pragma unroll
    for (int r = 0; r < 8; ++r) {             // D vgpr r -> sample m = mh + r
        const float plo = acc0[r] * acc0[r] + acc2[r] * acc2[r];   // p[n]
        const float phi = acc1[r] * acc1[r] + acc3[r] * acc3[r];   // p[n+16]
        Atile[wave][prow + mh + r][(n & 7)]     = (_Float16)plo;
        Atile[wave][prow + mh + r][(n & 7) + 8] = (_Float16)phi;
    }
    __syncthreads();

    // ---- 5th WMMA: ev[16 samples][5 wires] = P(16x32) x S(32x16) ----
    const v16h a2 = *(const v16h*)(&Atile[wave][lane][0]);
    v16h bs;                                  // sign matrix, built in registers
    const int g = lane >> 4;
    const int sh = (n < 5) ? (4 - n) : 0;     // bit position for wire n
#pragma unroll
    for (int e = 0; e < 16; ++e) {
        const int K = (e < 8) ? (g * 8 + e) : (16 + g * 8 + (e - 8));
        const float s = (n < 5) ? (((K >> sh) & 1) ? -1.f : 1.f) : 0.f;
        bs[e] = (_Float16)s;
    }
    v8f ev = __builtin_amdgcn_wmma_f32_16x16x32_f16(false, a2, false, bs, (short)0, z, false, false);

    // ---- stage expvals in LDS (reuse raw), then coalesced b128 stores ----
    float* evbuf = raw;                       // 640 floats per block
    if (n < 5) {
#pragma unroll
        for (int r = 0; r < 8; ++r)
            evbuf[(wave * 16 + mh + r) * 5 + n] = ev[r];
    }
    __syncthreads();
    if (tid < 160) {
        float4* o4 = (float4*)(out + (long long)blockIdx.x * 640);
        o4[tid] = ((const float4*)evbuf)[tid];
    }
}

// ---------------------------------------------------------------------------
extern "C" void kernel_launch(void* const* d_in, const int* in_sizes, int n_in,
                              void* d_out, int out_size, void* d_ws, size_t ws_size,
                              hipStream_t stream) {
    const float* in  = (const float*)d_in[0];   // (B, 30) float32
    const float* wts = (const float*)d_in[1];   // (4, 5, 3) float32
    _Float16* Bbuf   = (_Float16*)d_ws;         // 4*32*16 halves = 4 KB
    float* out       = (float*)d_out;           // (B, 5) float32

    const int B = in_sizes[0] / 30;             // 524288 (multiple of 128)

    build_unitary<<<1, 32, 0, stream>>>(wts, Bbuf);
    qsim_main<<<B / 128, 256, 0, stream>>>(in, Bbuf, out);
}
